// SwinTransformerEncoder_15564961481378
// MI455X (gfx1250) — matrix operand order repaired and checked by reading, not compiled, not run on hardware
//
#include <hip/hip_runtime.h>

// ---------------------------------------------------------------------------
// Swin Transformer encoder (2 blocks) for gfx1250 using v_wmma_f32_16x16x32_bf16
// H=W=64, C=256, heads=8 (d=32), window=8 (N=64 tokens), B=8, hidden=1024
// ---------------------------------------------------------------------------

typedef __attribute__((ext_vector_type(16))) __bf16 v16bf;
typedef __attribute__((ext_vector_type(8)))  __bf16 bf16x8;
typedef __attribute__((ext_vector_type(8)))  float  v8f;

#define CAT16(a,b) __builtin_shufflevector(a, b, 0,1,2,3,4,5,6,7,8,9,10,11,12,13,14,15)

__device__ inline __bf16 f2bf(float f) {
  unsigned u; __builtin_memcpy(&u, &f, 4);
  unsigned r = u + 0x7fffu + ((u >> 16) & 1u);   // round-to-nearest-even
  unsigned short h = (unsigned short)(r >> 16);
  __bf16 b; __builtin_memcpy(&b, &h, 2);
  return b;
}

__device__ inline v8f wmma_bf16(v16bf a, v16bf b, v8f c) {
  return __builtin_amdgcn_wmma_f32_16x16x32_bf16(false, a, false, b, (short)0, c, false, false);
}

// A-fragment: 16 contiguous K elements (lane half picks K offset 0 or 16)
__device__ inline v16bf ldfragA(const __bf16* p) {
  bf16x8 lo = *(const bf16x8*)p;
  bf16x8 hi = *(const bf16x8*)(p + 8);
  return CAT16(lo, hi);
}
// B-fragment from K-contiguous storage (W^T[n][k]): 16 contiguous bf16
__device__ inline v16bf ldfragB(const __bf16* p) {
  bf16x8 lo = *(const bf16x8*)p;
  bf16x8 hi = *(const bf16x8*)(p + 8);
  return CAT16(lo, hi);
}

// ---------------------------------------------------------------------------
// Weight convert + transpose: W f32 [K][N] -> Wt bf16 [N][K]
// ---------------------------------------------------------------------------
__global__ void convT_kernel(const float* __restrict__ W, __bf16* __restrict__ Wt,
                             int K, int N) {
  int idx = blockIdx.x * 256 + threadIdx.x;
  if (idx >= K * N) return;
  int k = idx / N, n = idx % N;
  Wt[(size_t)n * K + k] = f2bf(W[(size_t)k * N + n]);
}

// ---------------------------------------------------------------------------
// LayerNorm (256 ch), optional cyclic shift + window partition.
// shift < 0: identity token order.  shift >= 0: output token o is window-order,
// source = rolled-by-(-shift) image position. One wave per token.
// ---------------------------------------------------------------------------
__global__ void ln_kernel(const float* __restrict__ x, __bf16* __restrict__ out,
                          const float* __restrict__ g, const float* __restrict__ b,
                          int shift) {
  int l = threadIdx.x & 31;
  int wave = threadIdx.x >> 5;
  int o = blockIdx.x * 8 + wave;          // output token index [0, 32768)
  int src;
  if (shift < 0) {
    src = o;
  } else {
    int w = o >> 6, wt = o & 63;
    int b_ = w >> 6, wy = (w >> 3) & 7, wx = w & 7;
    int ty = wt >> 3, tx = wt & 7;
    int gy = (wy * 8 + ty + shift) & 63;
    int gx = (wx * 8 + tx + shift) & 63;
    src = b_ * 4096 + gy * 64 + gx;
  }
  const float* xp = x + (size_t)src * 256;
  float v[8];
  float s = 0.f, s2 = 0.f;
#pragma unroll
  for (int j = 0; j < 8; ++j) {
    int c = l + 32 * j;
    v[j] = xp[c];
    s += v[j]; s2 += v[j] * v[j];
  }
#pragma unroll
  for (int m = 1; m < 32; m <<= 1) {
    s  += __shfl_xor(s,  m);
    s2 += __shfl_xor(s2, m);
  }
  float mu  = s * (1.f / 256.f);
  float var = s2 * (1.f / 256.f) - mu * mu;
  float inv = rsqrtf(var + 1e-5f);
  __bf16* op = out + (size_t)o * 256;
#pragma unroll
  for (int j = 0; j < 8; ++j) {
    int c = l + 32 * j;
    op[c] = f2bf((v[j] - mu) * inv * g[c] + b[c]);
  }
}

// ---------------------------------------------------------------------------
// WMMA GEMM:  C[M][N] = A[M][K](bf16) * Wt[N][K]^T(bf16) + bias, epilogue EPI.
// Wave computes a 32(M) x 64(N) tile; K stepped by 32.  Double-buffered:
// next K-step's 6 fragments (2 A + 4 B) are loaded BEFORE the current WMMA
// burst, so the wait at each burst covers loads issued a full iteration ago.
//   EPI 0: qkv    -> bf16 out, scale q-part (n<256) by d^-1/2
//   EPI 1: proj   -> window-reverse + unshift + residual add into xio (f32)
//   EPI 2: fc1    -> exact GELU, bf16 out
//   EPI 3: fc2    -> residual add into xio (f32)
// ---------------------------------------------------------------------------
template <int EPI>
__global__ void gemm_bf16_kernel(const __bf16* __restrict__ A,
                                 const __bf16* __restrict__ Bt,
                                 const float* __restrict__ bias,
                                 __bf16* __restrict__ outbf,
                                 float* __restrict__ xio,
                                 int M, int N, int K, int shift) {
  int l    = threadIdx.x & 31;
  int wid  = blockIdx.x * (blockDim.x >> 5) + (threadIdx.x >> 5);
  int ntiles = N >> 6;
  int mt = wid / ntiles, nt = wid % ntiles;
  int m0 = mt << 5, n0 = nt << 6;
  int half = l >> 4, lm = l & 15;
  int aoff = half * 16, boff = half * 16;

  v8f acc[2][4] = {};
  const __bf16* ap0 = A + (size_t)(m0 + lm) * K + aoff;
  const __bf16* ap1 = A + (size_t)(m0 + 16 + lm) * K + aoff;
  const __bf16* bp0 = Bt + (size_t)(n0 + lm) * K + boff;
  const __bf16* bp1 = bp0 + (size_t)16 * K;
  const __bf16* bp2 = bp0 + (size_t)32 * K;
  const __bf16* bp3 = bp0 + (size_t)48 * K;

  // prologue: load first K-step's fragments
  v16bf a0 = ldfragA(ap0), a1 = ldfragA(ap1);
  v16bf b0 = ldfragB(bp0), b1 = ldfragB(bp1), b2 = ldfragB(bp2), b3 = ldfragB(bp3);

  for (int kk = 32; kk < K; kk += 32) {
    // issue next iteration's loads first (one clause, no wait yet)
    v16bf na0 = ldfragA(ap0 + kk), na1 = ldfragA(ap1 + kk);
    v16bf nb0 = ldfragB(bp0 + kk), nb1 = ldfragB(bp1 + kk);
    v16bf nb2 = ldfragB(bp2 + kk), nb3 = ldfragB(bp3 + kk);
    __builtin_prefetch(ap0 + kk + 32, 0, 1);
    __builtin_prefetch(ap1 + kk + 32, 0, 1);
    // WMMA burst on current fragments (loaded last iteration)
    acc[0][0] = wmma_bf16(a0, b0, acc[0][0]);
    acc[1][0] = wmma_bf16(a1, b0, acc[1][0]);
    acc[0][1] = wmma_bf16(a0, b1, acc[0][1]);
    acc[1][1] = wmma_bf16(a1, b1, acc[1][1]);
    acc[0][2] = wmma_bf16(a0, b2, acc[0][2]);
    acc[1][2] = wmma_bf16(a1, b2, acc[1][2]);
    acc[0][3] = wmma_bf16(a0, b3, acc[0][3]);
    acc[1][3] = wmma_bf16(a1, b3, acc[1][3]);
    a0 = na0; a1 = na1; b0 = nb0; b1 = nb1; b2 = nb2; b3 = nb3;
  }
  // epilogue K-step
  acc[0][0] = wmma_bf16(a0, b0, acc[0][0]);
  acc[1][0] = wmma_bf16(a1, b0, acc[1][0]);
  acc[0][1] = wmma_bf16(a0, b1, acc[0][1]);
  acc[1][1] = wmma_bf16(a1, b1, acc[1][1]);
  acc[0][2] = wmma_bf16(a0, b2, acc[0][2]);
  acc[1][2] = wmma_bf16(a1, b2, acc[1][2]);
  acc[0][3] = wmma_bf16(a0, b3, acc[0][3]);
  acc[1][3] = wmma_bf16(a1, b3, acc[1][3]);

#pragma unroll
  for (int u = 0; u < 2; ++u) {
#pragma unroll
    for (int t = 0; t < 4; ++t) {
#pragma unroll
      for (int r = 0; r < 8; ++r) {
        int n = n0 + 16 * t + lm;
        int m = m0 + u * 16 + half * 8 + r;
        float v = acc[u][t][r] + bias[n];
        if (EPI == 0) {
          if (n < 256) v *= 0.1767766953f;                 // q * d^-1/2
          outbf[(size_t)m * N + n] = f2bf(v);
        } else if (EPI == 1) {
          // window reverse + roll(+shift) + residual
          int w = m >> 6, wt = m & 63;
          int b_ = w >> 6, wy = (w >> 3) & 7, wx = w & 7;
          int ty = wt >> 3, tx = wt & 7;
          int gy = (wy * 8 + ty + shift) & 63;
          int gx = (wx * 8 + tx + shift) & 63;
          size_t pos = (size_t)b_ * 4096 + gy * 64 + gx;
          xio[pos * 256 + n] += v;
        } else if (EPI == 2) {
          float gl = 0.5f * v * (1.0f + erff(v * 0.70710678118f));
          outbf[(size_t)m * N + n] = f2bf(gl);
        } else {
          xio[(size_t)m * 256 + n] += v;
        }
      }
    }
  }
}

// ---------------------------------------------------------------------------
// Window attention: one block (4 waves, 128 thr) per (window, head).
// qkv layout: [token 32768][768] with 768 = {q|k|v} x 8 heads x 32 dims.
// Wave w handles rows m0 = 16*w of the 64x64 score matrix.
// MASKED is a compile-time flag (layer 1 shift mask), fully branchless.
// ---------------------------------------------------------------------------
template <int MASKED>
__global__ void attn_kernel(const __bf16* __restrict__ qkv,
                            __bf16* __restrict__ aout,
                            const float* __restrict__ rpb) {
  __shared__ alignas(16) __bf16 sVT[32 * 64];       // V^T [dim][token]
  __shared__ alignas(16) __bf16 sP[4][16 * 64];     // per-wave softmax probs

  int w = blockIdx.x >> 3;
  int h = blockIdx.x & 7;
  int tid = threadIdx.x;
  int l = tid & 31, wave = tid >> 5;
  int half = l >> 4, lm = l & 15;
  size_t base = (size_t)w * 64 * 768;

  // stage V^T: 128 threads x 16 contiguous elems
  {
    int tok = tid >> 1;
    int dbase = (tid & 1) * 16;
    const __bf16* vp = qkv + base + (size_t)tok * 768 + 512 + h * 32 + dbase;
#pragma unroll
    for (int j = 0; j < 16; ++j) sVT[(dbase + j) * 64 + tok] = vp[j];
  }
  __syncthreads();

  int m0 = wave * 16;

  // S = Q(16x32) @ K^T(32x64): load all fragments, then WMMA burst
  v8f s[4] = {};
  {
    const __bf16* qp = qkv + base + (size_t)(m0 + lm) * 768 + h * 32 + half * 16;
    v16bf af = ldfragA(qp);
    v16bf kf[4];
#pragma unroll
    for (int t = 0; t < 4; ++t)
      kf[t] = ldfragB(qkv + base + (size_t)(16 * t + lm) * 768 + 256 + h * 32 + half * 16);
#pragma unroll
    for (int t = 0; t < 4; ++t)
      s[t] = wmma_bf16(af, kf[t], s[t]);
  }

  // relative position bias (+ branchless shift mask)
  int wy = (w >> 3) & 7, wx = w & 7;
  const float* rpbh = rpb + h;
#pragma unroll
  for (int t = 0; t < 4; ++t) {
    int ni = 16 * t + lm;               // col token in window (per-lane const)
    int nyy = ni >> 3, nxx = ni & 7;
    int lab_n;
    if (MASKED) {
      int gy2 = wy * 8 + nyy, gx2 = wx * 8 + nxx;
      lab_n = (gy2 < 56 ? 0 : (gy2 < 60 ? 1 : 2)) * 3 +
              (gx2 < 56 ? 0 : (gx2 < 60 ? 1 : 2));
    }
#pragma unroll
    for (int r = 0; r < 8; ++r) {
      int mi = m0 + half * 8 + r;       // row token in window
      int dy = (mi >> 3) - nyy + 7;
      int dx = (mi & 7) - nxx + 7;
      float v = s[t][r] + rpbh[(dy * 15 + dx) * 8];
      if (MASKED) {
        int gy1 = wy * 8 + (mi >> 3), gx1 = wx * 8 + (mi & 7);
        int lab_m = (gy1 < 56 ? 0 : (gy1 < 60 ? 1 : 2)) * 3 +
                    (gx1 < 56 ? 0 : (gx1 < 60 ? 1 : 2));
        v += (lab_m != lab_n) ? -100.0f : 0.0f;   // v_cndmask, no branch
      }
      s[t][r] = v;
    }
  }

  // softmax per row (64-wide): local over t + xor-shuffle over 16-lane half
#pragma unroll
  for (int r = 0; r < 8; ++r) {
    float mx = s[0][r];
#pragma unroll
    for (int t = 1; t < 4; ++t) mx = fmaxf(mx, s[t][r]);
#pragma unroll
    for (int m = 1; m < 16; m <<= 1) mx = fmaxf(mx, __shfl_xor(mx, m));
    float sum = 0.f;
#pragma unroll
    for (int t = 0; t < 4; ++t) {
      float e = __expf(s[t][r] - mx);
      s[t][r] = e;
      sum += e;
    }
#pragma unroll
    for (int m = 1; m < 16; m <<= 1) sum += __shfl_xor(sum, m);
    float inv = 1.f / sum;
#pragma unroll
    for (int t = 0; t < 4; ++t) s[t][r] *= inv;
  }

  // P -> LDS in row-major [16][64] (A-fragment source)
#pragma unroll
  for (int t = 0; t < 4; ++t)
#pragma unroll
    for (int r = 0; r < 8; ++r)
      sP[wave][(half * 8 + r) * 64 + 16 * t + lm] = f2bf(s[t][r]);
  __syncthreads();

  // O = P(16x64) @ V(64x32): load all LDS fragments, then WMMA burst
  v8f o[2] = {};
  {
    v16bf pf[2], vf[2][2];
#pragma unroll
    for (int kx = 0; kx < 2; ++kx) {
      pf[kx] = ldfragA(&sP[wave][lm * 64 + kx * 32 + half * 16]);
#pragma unroll
      for (int t = 0; t < 2; ++t)
        vf[kx][t] = ldfragB(&sVT[(16 * t + lm) * 64 + kx * 32 + half * 16]);
    }
#pragma unroll
    for (int kx = 0; kx < 2; ++kx)
#pragma unroll
      for (int t = 0; t < 2; ++t)
        o[t] = wmma_bf16(pf[kx], vf[kx][t], o[t]);
  }

  // store in token-major [32768][256], channel = h*32 + dim
#pragma unroll
  for (int t = 0; t < 2; ++t)
#pragma unroll
    for (int r = 0; r < 8; ++r) {
      int mi = m0 + half * 8 + r;
      int dim = 16 * t + lm;
      aout[((size_t)w * 64 + mi) * 256 + h * 32 + dim] = f2bf(o[t][r]);
    }
}

// ---------------------------------------------------------------------------
// Host launcher
// ---------------------------------------------------------------------------
extern "C" void kernel_launch(void* const* d_in, const int* in_sizes, int n_in,
                              void* d_out, int out_size, void* d_ws, size_t ws_size,
                              hipStream_t stream) {
  const float* x      = (const float*)d_in[0];
  const float* qkv_w  = (const float*)d_in[1];
  const float* qkv_b  = (const float*)d_in[2];
  const float* proj_w = (const float*)d_in[3];
  const float* proj_b = (const float*)d_in[4];
  const float* ln1_g  = (const float*)d_in[5];
  const float* ln1_b  = (const float*)d_in[6];
  const float* ln2_g  = (const float*)d_in[7];
  const float* ln2_b  = (const float*)d_in[8];
  const float* fc1_w  = (const float*)d_in[9];
  const float* fc1_b  = (const float*)d_in[10];
  const float* fc2_w  = (const float*)d_in[11];
  const float* fc2_b  = (const float*)d_in[12];
  const float* rpb    = (const float*)d_in[13];

  char* ws = (char*)d_ws;
  __bf16* qkvT  = (__bf16*)ws;                                       // 768*256
  __bf16* projT = (__bf16*)(ws + 393216);                            // 256*256
  __bf16* fc1T  = (__bf16*)(ws + 393216 + 131072);                   // 1024*256
  __bf16* fc2T  = (__bf16*)(ws + 393216 + 131072 + 524288);          // 256*1024
  size_t oA = 1572864;
  __bf16* qkvbuf  = (__bf16*)(ws + oA);              // 32768*768 bf16 (50.3MB)
  __bf16* hidbuf  = qkvbuf;                          // reuse: 32768*1024 bf16
  size_t oB = oA + 67108864;
  __bf16* hwbuf   = (__bf16*)(ws + oB);              // 32768*256 bf16
  __bf16* ln2buf  = hwbuf;                           // reuse
  size_t oC = oB + 16777216;
  __bf16* attnout = (__bf16*)(ws + oC);              // 32768*256 bf16
  float* xio = (float*)d_out;

  // x lives in d_out for both layers (residuals updated in place)
  hipMemcpyAsync(xio, x, (size_t)8 * 4096 * 256 * sizeof(float),
                 hipMemcpyDeviceToDevice, stream);

  for (int i = 0; i < 2; ++i) {
    int shift = (i & 1) ? 4 : 0;

    convT_kernel<<<(256 * 768 + 255) / 256, 256, 0, stream>>>(qkv_w + (size_t)i * 256 * 768, qkvT, 256, 768);
    convT_kernel<<<(256 * 256 + 255) / 256, 256, 0, stream>>>(proj_w + (size_t)i * 256 * 256, projT, 256, 256);
    convT_kernel<<<(256 * 1024 + 255) / 256, 256, 0, stream>>>(fc1_w + (size_t)i * 256 * 1024, fc1T, 256, 1024);
    convT_kernel<<<(1024 * 256 + 255) / 256, 256, 0, stream>>>(fc2_w + (size_t)i * 1024 * 256, fc2T, 1024, 256);

    // LN1 + shift + window partition -> bf16
    ln_kernel<<<4096, 256, 0, stream>>>(xio, hwbuf, ln1_g + i * 256, ln1_b + i * 256, shift);

    // qkv GEMM: [32768,256] x [256,768]  (1024 M-tiles x 12 N-tiles)
    gemm_bf16_kernel<0><<<3072, 128, 0, stream>>>(hwbuf, qkvT, qkv_b + i * 768,
                                                  qkvbuf, nullptr, 32768, 768, 256, 0);

    // window attention (512 windows x 8 heads)
    if (i == 0)
      attn_kernel<0><<<4096, 128, 0, stream>>>(qkvbuf, attnout, rpb);
    else
      attn_kernel<1><<<4096, 128, 0, stream>>>(qkvbuf, attnout, rpb + 225 * 8);

    // proj GEMM + window reverse + residual  (1024 x 4 tiles)
    gemm_bf16_kernel<1><<<1024, 128, 0, stream>>>(attnout, projT, proj_b + i * 256,
                                                  nullptr, xio, 32768, 256, 256, shift);

    // LN2 -> bf16
    ln_kernel<<<4096, 256, 0, stream>>>(xio, ln2buf, ln2_g + i * 256, ln2_b + i * 256, -1);

    // fc1 GEMM + GELU  (1024 x 16 tiles)
    gemm_bf16_kernel<2><<<4096, 128, 0, stream>>>(ln2buf, fc1T, fc1_b + i * 1024,
                                                  hidbuf, nullptr, 32768, 1024, 256, 0);

    // fc2 GEMM + residual  (1024 x 4 tiles)
    gemm_bf16_kernel<3><<<1024, 128, 0, stream>>>(hidbuf, fc2T, fc2_b + i * 256,
                                                  nullptr, xio, 32768, 256, 1024, 0);
  }
}
